// basket_GRU_6399501271095
// MI455X (gfx1250) — matrix operand order repaired and checked by reading, not compiled, not run on hardware
//
#include <hip/hip_runtime.h>
#include <hip/hip_bf16.h>
#include <math.h>

// ---- problem constants ----
#define BB 128
#define TT 100
#define DD 4096
#define HH 256
#define MROWS (BB * TT)      // 12800
#define N3H   (3 * HH)       // 768

typedef _Float16 v8h  __attribute__((ext_vector_type(8)));
typedef _Float16 v16h __attribute__((ext_vector_type(16)));
typedef float    v8f  __attribute__((ext_vector_type(8)));

// ---------------------------------------------------------------------------
// WMMA helpers (wave32, V_WMMA_F32_16X16X32_F16)
// A fragment (16xK tile, f16, row-major src, ld = row stride in elements):
//   lane L: M = L&15, group g = L>>4
//   halves 0..7  <- K = k0 + g*8 + 0..7       (VGPRs 0..3)
//   halves 8..15 <- K = k0 + 16 + g*8 + 0..7  (VGPRs 4..7)
// B fragment (Kx16 tile, f16, row-major src):
//   lane L: N = L&15, g = L>>4; halves 0..7 <- K = g*8 + 0..7, halves 8..15 <- K = 16 + g*8 + 0..7
// C/D (16x16 f32): vgpr e: M = e + 8*(L>>4), N = L&15
// ---------------------------------------------------------------------------
__device__ __forceinline__ v16h load_frag_a(const _Float16* A, int lda, int lane) {
    const int g = lane >> 4, m = lane & 15;
    const _Float16* p = A + (size_t)m * lda + g * 8;
    v8h lo = *(const v8h*)(p);
    v8h hi = *(const v8h*)(p + 16);
    v16h r;
#pragma unroll
    for (int i = 0; i < 8; ++i) { r[i] = lo[i]; r[i + 8] = hi[i]; }
    return r;
}

__device__ __forceinline__ v16h load_frag_b(const _Float16* Bp, int ldb, int lane) {
    const int g = lane >> 4, n = lane & 15;
    const _Float16* p = Bp + n;
    v16h r;
#pragma unroll
    for (int i = 0; i < 8; ++i) { r[i] = p[(size_t)(g * 8 + i) * ldb]; r[i + 8] = p[(size_t)(16 + g * 8 + i) * ldb]; }
    return r;
}

__device__ __forceinline__ v8f wmma16(v16h a, v16h b, v8f c) {
    return __builtin_amdgcn_wmma_f32_16x16x32_f16(false, a, false, b, (short)0, c,
                                                  false, false);
}

__device__ __forceinline__ float sigmoidf_(float x) {
    return 1.0f / (1.0f + __expf(-x));
}

// ---------------------------------------------------------------------------
// Prep kernels: f32 -> f16 conversions and weight packing
// ---------------------------------------------------------------------------
__global__ void k_cvt_f16(const float* __restrict__ s, _Float16* __restrict__ d, long n) {
    long i = (long)blockIdx.x * blockDim.x + threadIdx.x;
    long stride = (long)gridDim.x * blockDim.x;
    for (; i < n; i += stride) d[i] = (_Float16)s[i];
}

__global__ void k_pack_wx(const float* __restrict__ Wfx, const float* __restrict__ Wmx,
                          const float* __restrict__ Wcx, _Float16* __restrict__ dst) {
    long i = (long)blockIdx.x * blockDim.x + threadIdx.x;
    if (i >= (long)DD * N3H) return;
    int dcol = (int)(i % N3H);
    long drow = i / N3H;
    float v;
    if (dcol < HH)            v = Wfx[drow * HH + dcol];
    else if (dcol < 2 * HH)   v = Wmx[drow * HH + (dcol - HH)];
    else                      v = Wcx[drow * HH + (dcol - 2 * HH)];
    dst[i] = (_Float16)v;
}

__global__ void k_pack_bias(const float* __restrict__ bf, const float* __restrict__ bm,
                            const float* __restrict__ bc, float* __restrict__ dst) {
    int i = blockIdx.x * blockDim.x + threadIdx.x;
    if (i >= N3H) return;
    dst[i] = (i < HH) ? bf[i] : (i < 2 * HH) ? bm[i - HH] : bc[i - 2 * HH];
}

// ---------------------------------------------------------------------------
// GEMM 1: xproj[12800, 768] = x_f16[12800,4096] @ Wxcat_f16[4096,768] + bcat
// Block = 256 threads (8 waves) covering a 128x64 tile.
// Each wave: 2x2 register blocking (32x32 patch) -> 1 fragment load per wmma.
// ---------------------------------------------------------------------------
__global__ __launch_bounds__(256) void k_gemm_xproj(
    const _Float16* __restrict__ Xh, const _Float16* __restrict__ Wx,
    const float* __restrict__ bcat, float* __restrict__ xproj) {
    const int lane  = threadIdx.x & 31;
    const int wave  = threadIdx.x >> 5;
    const int waveM = wave >> 1;          // 0..3
    const int waveN = wave & 1;           // 0..1
    const int m0 = blockIdx.x * 128 + waveM * 32;
    const int n0 = blockIdx.y * 64 + waveN * 32;

    const _Float16* A0 = Xh + (size_t)m0 * DD;
    const _Float16* A1 = Xh + (size_t)(m0 + 16) * DD;
    v8f acc00 = {}, acc01 = {}, acc10 = {}, acc11 = {};
    for (int k = 0; k < DD; k += 32) {
        __builtin_prefetch(A0 + k + 128, 0, 0);
        __builtin_prefetch(A1 + k + 128, 0, 0);
        v16h a0 = load_frag_a(A0 + k, DD, lane);
        v16h a1 = load_frag_a(A1 + k, DD, lane);
        v16h b0 = load_frag_b(Wx + (size_t)k * N3H + n0, N3H, lane);
        v16h b1 = load_frag_b(Wx + (size_t)k * N3H + n0 + 16, N3H, lane);
        acc00 = wmma16(a0, b0, acc00);
        acc01 = wmma16(a0, b1, acc01);
        acc10 = wmma16(a1, b0, acc10);
        acc11 = wmma16(a1, b1, acc11);
    }
    const int g = lane >> 4, N = lane & 15;
    const float bv0 = bcat[n0 + N];
    const float bv1 = bcat[n0 + 16 + N];
#pragma unroll
    for (int e = 0; e < 8; ++e) {
        int r0 = m0 + e + 8 * g;
        int r1 = r0 + 16;
        xproj[(size_t)r0 * N3H + n0 + N]      = acc00[e] + bv0;
        xproj[(size_t)r0 * N3H + n0 + 16 + N] = acc01[e] + bv1;
        xproj[(size_t)r1 * N3H + n0 + N]      = acc10[e] + bv0;
        xproj[(size_t)r1 * N3H + n0 + 16 + N] = acc11[e] + bv1;
    }
}

// ---------------------------------------------------------------------------
// Recurrence: 8 blocks x 16 batch rows, 100 sequential steps.
// h kept in LDS (f32 master + f16 staging for WMMA A fragments).
// Each wave owns 2 N-tiles (32 hidden units). Weights stay hot in L2.
// ---------------------------------------------------------------------------
__global__ __launch_bounds__(256) void k_recurrence(
    const float* __restrict__ xproj,
    const _Float16* __restrict__ Whf, const _Float16* __restrict__ Whm,
    const _Float16* __restrict__ Whc, _Float16* __restrict__ hid) {
    __shared__ __align__(32) float     h32[16][HH];
    __shared__ __align__(32) _Float16  h16[16][HH];
    __shared__ __align__(32) _Float16  hm16[16][HH];

    const int lane = threadIdx.x & 31;
    const int wave = threadIdx.x >> 5;
    const int b0 = blockIdx.x * 16;
    const int g = lane >> 4, N = lane & 15;

    for (int i = threadIdx.x; i < 16 * HH; i += 256) {
        (&h32[0][0])[i] = 0.0f;
        (&h16[0][0])[i] = (_Float16)0.0f;
    }
    __syncthreads();

    for (int t = 0; t < TT; ++t) {
        // --- fused GEMM: h @ W_fh and h @ W_mh ---
        v8f aF[2] = {v8f{}, v8f{}};
        v8f aM[2] = {v8f{}, v8f{}};
        for (int k = 0; k < HH; k += 32) {
            v16h a = load_frag_a(&h16[0][k], HH, lane);
#pragma unroll
            for (int s = 0; s < 2; ++s) {
                int n0 = wave * 32 + s * 16;
                v16h bf = load_frag_b(Whf + (size_t)k * HH + n0, HH, lane);
                aF[s] = wmma16(a, bf, aF[s]);
                v16h bm = load_frag_b(Whm + (size_t)k * HH + n0, HH, lane);
                aM[s] = wmma16(a, bm, aM[s]);
            }
        }
        // --- modul = sigmoid(xm + h@W_mh); hm = h * modul ---
#pragma unroll
        for (int s = 0; s < 2; ++s) {
            int nn = wave * 32 + s * 16 + N;
#pragma unroll
            for (int e = 0; e < 8; ++e) {
                int m = e + 8 * g;
                size_t row = (size_t)(b0 + m) * TT + t;
                float xm = xproj[row * N3H + HH + nn];
                float mo = sigmoidf_(xm + aM[s][e]);
                hm16[m][nn] = (_Float16)(h32[m][nn] * mo);
            }
        }
        __syncthreads();
        // --- GEMM: (h*modul) @ W_ch ---
        v8f aC[2] = {v8f{}, v8f{}};
        for (int k = 0; k < HH; k += 32) {
            v16h a = load_frag_a(&hm16[0][k], HH, lane);
#pragma unroll
            for (int s = 0; s < 2; ++s) {
                int n0 = wave * 32 + s * 16;
                v16h bc = load_frag_b(Whc + (size_t)k * HH + n0, HH, lane);
                aC[s] = wmma16(a, bc, aC[s]);
            }
        }
        // --- gate math + hidden update ---
#pragma unroll
        for (int s = 0; s < 2; ++s) {
            int nn = wave * 32 + s * 16 + N;
#pragma unroll
            for (int e = 0; e < 8; ++e) {
                int m = e + 8 * g;
                size_t row = (size_t)(b0 + m) * TT + t;
                float xf = xproj[row * N3H + nn];
                float xc = xproj[row * N3H + 2 * HH + nn];
                float f = sigmoidf_(xf + aF[s][e]);
                float cand = tanhf(xc + aC[s][e]);
                float hn = h32[m][nn] * (1.0f - f) + f * cand;
                h32[m][nn] = hn;
                h16[m][nn] = (_Float16)hn;
                hid[row * HH + nn] = (_Float16)hn;
            }
        }
        __syncthreads();
    }
}

// ---------------------------------------------------------------------------
// GEMM 2: out[12800,4096] = sigmoid(hid_f16[12800,256] @ Wph_f16[256,4096] + b_p)
// Same 2x2 register blocking: block tile 128x64, K = 256 (8 steps, unrolled).
// ---------------------------------------------------------------------------
__global__ __launch_bounds__(256) void k_gemm_out(
    const _Float16* __restrict__ Hh, const _Float16* __restrict__ Wph,
    const float* __restrict__ bp, float* __restrict__ out) {
    const int lane  = threadIdx.x & 31;
    const int wave  = threadIdx.x >> 5;
    const int waveM = wave >> 1;
    const int waveN = wave & 1;
    const int m0 = blockIdx.x * 128 + waveM * 32;
    const int n0 = blockIdx.y * 64 + waveN * 32;

    const _Float16* A0 = Hh + (size_t)m0 * HH;
    const _Float16* A1 = Hh + (size_t)(m0 + 16) * HH;
    v8f acc00 = {}, acc01 = {}, acc10 = {}, acc11 = {};
#pragma unroll
    for (int k = 0; k < HH; k += 32) {
        v16h a0 = load_frag_a(A0 + k, HH, lane);
        v16h a1 = load_frag_a(A1 + k, HH, lane);
        v16h b0 = load_frag_b(Wph + (size_t)k * DD + n0, DD, lane);
        v16h b1 = load_frag_b(Wph + (size_t)k * DD + n0 + 16, DD, lane);
        acc00 = wmma16(a0, b0, acc00);
        acc01 = wmma16(a0, b1, acc01);
        acc10 = wmma16(a1, b0, acc10);
        acc11 = wmma16(a1, b1, acc11);
    }
    const int g = lane >> 4, N = lane & 15;
    const float bv0 = bp[n0 + N];
    const float bv1 = bp[n0 + 16 + N];
#pragma unroll
    for (int e = 0; e < 8; ++e) {
        int r0 = m0 + e + 8 * g;
        int r1 = r0 + 16;
        out[(size_t)r0 * DD + n0 + N]      = sigmoidf_(acc00[e] + bv0);
        out[(size_t)r0 * DD + n0 + 16 + N] = sigmoidf_(acc01[e] + bv1);
        out[(size_t)r1 * DD + n0 + N]      = sigmoidf_(acc10[e] + bv0);
        out[(size_t)r1 * DD + n0 + 16 + N] = sigmoidf_(acc11[e] + bv1);
    }
}

// ---------------------------------------------------------------------------
// host launcher
// ---------------------------------------------------------------------------
extern "C" void kernel_launch(void* const* d_in, const int* in_sizes, int n_in,
                              void* d_out, int out_size, void* d_ws, size_t ws_size,
                              hipStream_t stream) {
    const float* x    = (const float*)d_in[0];
    const float* W_fx = (const float*)d_in[1];
    const float* W_fh = (const float*)d_in[2];
    const float* b_f  = (const float*)d_in[3];
    const float* W_mx = (const float*)d_in[4];
    const float* W_mh = (const float*)d_in[5];
    const float* b_m  = (const float*)d_in[6];
    const float* W_cx = (const float*)d_in[7];
    const float* W_ch = (const float*)d_in[8];
    const float* b_c  = (const float*)d_in[9];
    const float* W_ph = (const float*)d_in[10];
    const float* b_p  = (const float*)d_in[11];
    float* out = (float*)d_out;
    char* ws = (char*)d_ws;

    // workspace carve-up (256B aligned slabs)
    size_t cur = 0;
    auto take = [&](size_t bytes) -> char* {
        char* p = ws + cur;
        cur = (cur + bytes + 255) & ~(size_t)255;
        return p;
    };
    const long NX = (long)MROWS * DD;                      // 52,428,800
    _Float16* xh    = (_Float16*)take((size_t)NX * 2);     // x in f16 (105MB, L2-resident)
    _Float16* wxcat = (_Float16*)take((size_t)DD * N3H * 2);
    _Float16* whf   = (_Float16*)take((size_t)HH * HH * 2);
    _Float16* whm   = (_Float16*)take((size_t)HH * HH * 2);
    _Float16* whc   = (_Float16*)take((size_t)HH * HH * 2);
    _Float16* wph   = (_Float16*)take((size_t)HH * DD * 2);
    float*    bcat  = (float*)take((size_t)N3H * 4);
    float*    xproj = (float*)take((size_t)MROWS * N3H * 4);
    _Float16* hid   = (_Float16*)take((size_t)MROWS * HH * 2);

    // --- prep: conversions/packing ---
    {
        int blk = 256;
        long nx = NX;
        int gx = (int)((nx + blk * 8 - 1) / (blk * 8));    // 8 elems/thread grid-stride
        k_cvt_f16<<<gx, blk, 0, stream>>>(x, xh, nx);
        k_cvt_f16<<<(HH * HH + 255) / 256, blk, 0, stream>>>(W_fh, whf, (long)HH * HH);
        k_cvt_f16<<<(HH * HH + 255) / 256, blk, 0, stream>>>(W_mh, whm, (long)HH * HH);
        k_cvt_f16<<<(HH * HH + 255) / 256, blk, 0, stream>>>(W_ch, whc, (long)HH * HH);
        k_cvt_f16<<<(HH * DD + 255) / 256, blk, 0, stream>>>(W_ph, wph, (long)HH * DD);
        long nwx = (long)DD * N3H;
        k_pack_wx<<<(int)((nwx + 255) / 256), blk, 0, stream>>>(W_fx, W_mx, W_cx, wxcat);
        k_pack_bias<<<(N3H + 255) / 256, blk, 0, stream>>>(b_f, b_m, b_c, bcat);
    }

    // --- GEMM 1: input projections (fused f/m/c) ---
    k_gemm_xproj<<<dim3(MROWS / 128, N3H / 64), 256, 0, stream>>>(xh, wxcat, bcat, xproj);

    // --- recurrence over T ---
    k_recurrence<<<BB / 16, 256, 0, stream>>>(xproj, whf, whm, whc, hid);

    // --- GEMM 2: output projection + sigmoid ---
    k_gemm_out<<<dim3(MROWS / 128, DD / 64), 256, 0, stream>>>(hid, wph, b_p, out);
}
